// PredRNN_22539988369572
// MI455X (gfx1250) — matrix operand order repaired and checked by reading, not compiled, not run on hardware
//
#include <hip/hip_runtime.h>

// ---------------------------------------------------------------------------
// PredRNN forward for MI455X (gfx1250, wave32, WMMA bf16).
// B=16 T=20 H=W=32 C=16 L=4 NH=64 F=5 IL=10
// Convs as implicit GEMM on v_wmma_f32_16x16x32_bf16.
// Round 2: 32 rows/wave (2 A frags -> 8 wmma per B-frag set), async-to-LDS
// staging via global_load_async_to_lds_b128 + s_wait_asynccnt.
// ---------------------------------------------------------------------------

#define USE_ASYNC_LDS 1

typedef __bf16 bf16_t;
typedef __attribute__((ext_vector_type(16))) __bf16 v16bf;
typedef __attribute__((ext_vector_type(8)))  __bf16 v8bf;
typedef __attribute__((ext_vector_type(8)))  float  v8f;

#define NB   16        // batch
#define NT   20        // total length
#define NHW  1024      // 32*32
#define PIX  16384     // NB*NHW
#define NCH  16        // frame channels
#define NL   4         // layers
#define NHID 64        // hidden
#define NIL  10        // input length
#define NSTEP 19       // T-1

__device__ __forceinline__ bf16_t f2b(float x) { return (bf16_t)x; }
__device__ __forceinline__ float sigm(float x) { return 1.0f / (1.0f + __expf(-x)); }

// ---------------------------------------------------------------------------
// Weight prep: f32 [Cout][Cin][ks][ks] -> bf16 [ks*ks][CoutP][CinP], zero pad.
// ---------------------------------------------------------------------------
__global__ void wprep_kernel(const float* __restrict__ src, bf16_t* __restrict__ dst,
                             int Cout, int Cin, int CoutP, int CinP, int ks)
{
    int idx = blockIdx.x * 256 + threadIdx.x;
    int per = CoutP * CinP;
    int total = ks * ks * per;
    if (idx >= total) return;
    int t  = idx / per;
    int r  = idx - t * per;
    int co = r / CinP;
    int ci = r - co * CinP;
    float v = 0.0f;
    if (co < Cout && ci < Cin) {
        int ty = t / ks, tx = t - ty * ks;
        v = src[(((size_t)co * Cin + ci) * ks + ty) * ks + tx];
    }
    dst[idx] = f2b(v);
}

// ---------------------------------------------------------------------------
// Implicit-GEMM conv (ks=5 pad=2, or ks=1 pad=0).
//   act : bf16 NHWC  [PIX][CinP]          (channel fastest)
//   wt  : bf16       [ks*ks][CoutP][CinP]
//   out : f32        [CoutP][PIX]         (channel-major)
// Block: 256 thr = 8 waves. Tile: 256 pixels x 64 cout.
// Each wave: 32 rows (2 A frags), 2x4 v8f accumulators.
// ---------------------------------------------------------------------------
__global__ __launch_bounds__(256)
void conv_gemm(const bf16_t* __restrict__ act, const bf16_t* __restrict__ wt,
               float* __restrict__ out, int CinP, int CoutP, int ks)
{
    __shared__ bf16_t alds[12 * 36 * 32];   // [row][col][cin32] halo tile, 27KB

    const int tid  = threadIdx.x;
    const int wave = tid >> 5;
    const int lane = tid & 31;

    const int pixbase  = blockIdx.x * 256;      // 8 image rows of one sample
    const int coutbase = blockIdx.y * 64;
    const int b  = pixbase >> 10;
    const int y0 = (pixbase >> 5) & 31;

    const int pad   = ks >> 1;
    const int rows  = 8 + 2 * pad;
    const int cols  = 32 + 2 * pad;
    const int taps  = ks * ks;
    const int chunks = CinP >> 5;

    v8f acc[2][4] = {};

    const int kb = (lane & 16) ? 8 : 0;          // A frag K base (ISA layout)

    for (int cc = 0; cc < chunks; ++cc) {
        __syncthreads();   // previous chunk's reads done before restage
        // ---- stage input tile for this 32-channel slice (zero halo) ----
        for (int rc = tid; rc < rows * cols; rc += 256) {
            int r  = rc / cols;
            int ci = rc - r * cols;
            int gy = y0 - pad + r;
            int gx = ci - pad;
            bool ok = ((unsigned)gy < 32u) && ((unsigned)gx < 32u);
            bf16_t* dst = alds + rc * 32;
            if (ok) {
                const bf16_t* src = act +
                    ((size_t)((b << 10) + gy * 32 + gx) * CinP + (cc << 5));
#if USE_ASYNC_LDS
                // ASYNCcnt-tracked DMA into LDS, no VGPR round-trip.
                // LDS byte offset = low 32 bits of the generic address
                // (ISA 10.2, LDS aperture).
                unsigned int  lo = (unsigned int)(unsigned long long)dst;
                unsigned long long ga = (unsigned long long)src;
                #pragma unroll
                for (int j = 0; j < 4; ++j) {
                    asm volatile("global_load_async_to_lds_b128 %0, %1, off"
                                 :: "v"(lo + 16u * j), "v"(ga + 16ull * j)
                                 : "memory");
                }
#else
                const int4* s4 = (const int4*)src;
                int4* d4 = (int4*)dst;
                #pragma unroll
                for (int j = 0; j < 4; ++j) d4[j] = s4[j];
#endif
            } else {
                int4 z = make_int4(0, 0, 0, 0);
                int4* d4 = (int4*)dst;
                #pragma unroll
                for (int j = 0; j < 4; ++j) d4[j] = z;
            }
        }
#if USE_ASYNC_LDS
        asm volatile("s_wait_asynccnt 0" ::: "memory");
#endif
        // hint this chunk's B-matrix weights toward the caches
        __builtin_prefetch(wt + ((size_t)coutbase * CinP + (cc << 5)), 0, 1);
        __syncthreads();

        // ---- 25 (or 1) taps of K=32 GEMM-accumulate ----
        for (int t = 0; t < taps; ++t) {
            int dy = t / ks;
            int dx = t - dy * ks;

            v16bf afrag[2];
            #pragma unroll
            for (int mf = 0; mf < 2; ++mf) {
                int m = wave * 32 + mf * 16 + (lane & 15);
                const bf16_t* ap = alds +
                    ((((m >> 5) + dy) * cols + (m & 31) + dx) << 5) + kb;
                v8bf alo = *(const v8bf*)ap;          // K kb..kb+7
                v8bf ahi = *(const v8bf*)(ap + 16);   // K kb+16..kb+23
                afrag[mf] = __builtin_shufflevector(alo, ahi,
                    0,1,2,3,4,5,6,7,8,9,10,11,12,13,14,15);
            }

            const bf16_t* wb = wt +
                ((size_t)(t * CoutP + coutbase + (lane & 15)) * CinP) +
                (cc << 5) + ((lane & 16) ? 8 : 0);
            #pragma unroll
            for (int nt = 0; nt < 4; ++nt) {
                const bf16_t* bp = wb + (size_t)(nt * 16) * CinP;
                v16bf bfrag = __builtin_shufflevector(*(const v8bf*)bp, *(const v8bf*)(bp + 16),
                    0,1,2,3,4,5,6,7,8,9,10,11,12,13,14,15);
                acc[0][nt] = __builtin_amdgcn_wmma_f32_16x16x32_bf16(
                    false, afrag[0], false, bfrag, (short)0, acc[0][nt], false, false);
                acc[1][nt] = __builtin_amdgcn_wmma_f32_16x16x32_bf16(
                    false, afrag[1], false, bfrag, (short)0, acc[1][nt], false, false);
            }
        }
    }

    // ---- store D (f32 16x16 layout: VGPR v -> M=v / v+8; N = lane%16) ----
    #pragma unroll
    for (int mf = 0; mf < 2; ++mf) {
        #pragma unroll
        for (int nt = 0; nt < 4; ++nt) {
            int n = coutbase + nt * 16 + (lane & 15);
            float* op = out + (size_t)n * PIX + pixbase +
                        wave * 32 + mf * 16 + ((lane & 16) ? 8 : 0);
            #pragma unroll
            for (int v = 0; v < 8; ++v) op[v] = acc[mf][nt][v];
        }
    }
}

// ---------------------------------------------------------------------------
// LayerNorm over [Cg, H, W] per sample, in place on [Cg][PIX] channel-major.
// grid = NB samples, block = 1024.
// ---------------------------------------------------------------------------
__global__ __launch_bounds__(1024)
void ln_kernel(float* __restrict__ x, const float* __restrict__ gg,
               const float* __restrict__ bb, int Cg)
{
    __shared__ float s1[1024];
    __shared__ float s2[1024];
    const int b = blockIdx.x;
    const int tid = threadIdx.x;
    const int N = Cg << 10;

    float sum = 0.0f, sq = 0.0f;
    for (int i = tid; i < N; i += 1024) {
        int ch = i >> 10, p = i & 1023;
        float v = x[(size_t)ch * PIX + (b << 10) + p];
        sum += v; sq += v * v;
    }
    s1[tid] = sum; s2[tid] = sq;
    __syncthreads();
    for (int s = 512; s > 0; s >>= 1) {
        if (tid < s) { s1[tid] += s1[tid + s]; s2[tid] += s2[tid + s]; }
        __syncthreads();
    }
    float inv = 1.0f / (float)N;
    float mu  = s1[0] * inv;
    float var = s2[0] * inv - mu * mu;
    float rs  = rsqrtf(var + 1e-5f);
    for (int i = tid; i < N; i += 1024) {
        int ch = i >> 10, p = i & 1023;
        size_t xi = (size_t)ch * PIX + (b << 10) + p;
        x[xi] = (x[xi] - mu) * rs * gg[i] + bb[i];
    }
}

// ---------------------------------------------------------------------------
// Gate fusion: c_new, m_new; emits f32 states + bf16 acts (m, [c|m] 128ch).
// ---------------------------------------------------------------------------
__global__ void gates_kernel(const float* __restrict__ xcc, const float* __restrict__ hcc,
                             const float* __restrict__ mcc,
                             float* __restrict__ cst, float* __restrict__ mst,
                             bf16_t* __restrict__ actm, bf16_t* __restrict__ actmem2)
{
    int idx = blockIdx.x * 256 + threadIdx.x;
    if (idx >= NHID * PIX) return;
    int ch  = idx >> 14;
    int pix = idx & (PIX - 1);
    size_t o = (size_t)ch * PIX + pix;

    float i_x  = xcc[(size_t)(0 * NHID + ch) * PIX + pix];
    float f_x  = xcc[(size_t)(1 * NHID + ch) * PIX + pix];
    float g_x  = xcc[(size_t)(2 * NHID + ch) * PIX + pix];
    float ip_x = xcc[(size_t)(3 * NHID + ch) * PIX + pix];
    float fp_x = xcc[(size_t)(4 * NHID + ch) * PIX + pix];
    float gp_x = xcc[(size_t)(5 * NHID + ch) * PIX + pix];
    float i_h  = hcc[(size_t)(0 * NHID + ch) * PIX + pix];
    float f_h  = hcc[(size_t)(1 * NHID + ch) * PIX + pix];
    float g_h  = hcc[(size_t)(2 * NHID + ch) * PIX + pix];
    float i_m  = mcc[(size_t)(0 * NHID + ch) * PIX + pix];
    float f_m  = mcc[(size_t)(1 * NHID + ch) * PIX + pix];
    float g_m  = mcc[(size_t)(2 * NHID + ch) * PIX + pix];

    float co = cst[o], mo = mst[o];
    float it = sigm(i_x + i_h);
    float ft = sigm(f_x + f_h);
    float gt = tanhf(g_x + g_h);
    float cn = ft * co + it * gt;
    float itp = sigm(ip_x + i_m);
    float ftp = sigm(fp_x + f_m);
    float gtp = tanhf(gp_x + g_m);
    float mn  = ftp * mo + itp * gtp;

    cst[o] = cn;
    mst[o] = mn;
    actm[(size_t)pix * NHID + ch]            = f2b(mn);
    actmem2[(size_t)pix * 128 + ch]          = f2b(cn);
    actmem2[(size_t)pix * 128 + NHID + ch]   = f2b(mn);
}

// ---------------------------------------------------------------------------
// Output gate: h_new = sigmoid(o_x + o_h + occ) * tanh(ccc) -> bf16 act_h.
// ---------------------------------------------------------------------------
__global__ void outgate_kernel(const float* __restrict__ xcc, const float* __restrict__ hcc,
                               const float* __restrict__ occ, const float* __restrict__ ccc,
                               bf16_t* __restrict__ acth)
{
    int idx = blockIdx.x * 256 + threadIdx.x;
    if (idx >= NHID * PIX) return;
    int ch  = idx >> 14;
    int pix = idx & (PIX - 1);
    float ox = xcc[(size_t)(6 * NHID + ch) * PIX + pix];
    float oh = hcc[(size_t)(3 * NHID + ch) * PIX + pix];
    float ot = sigm(ox + oh + occ[(size_t)ch * PIX + pix]);
    float hn = ot * tanhf(ccc[(size_t)ch * PIX + pix]);
    acth[(size_t)pix * NHID + ch] = f2b(hn);
}

// ---------------------------------------------------------------------------
// Input construction: net = frame (t<IL) else mask*frame + (1-mask)*x_gen.
// ---------------------------------------------------------------------------
__global__ void net_kernel(const float* __restrict__ frames, const float* __restrict__ mask,
                           const float* __restrict__ xgen, bf16_t* __restrict__ actnet, int t)
{
    int idx = blockIdx.x * 256 + threadIdx.x;
    if (idx >= PIX * NCH) return;
    int pix = idx >> 4;
    int c   = idx & 15;
    int b   = pix >> 10;
    int p   = pix & 1023;
    float fr = frames[(((size_t)b * NT + t) * NHW + p) * NCH + c];
    float v;
    if (t < NIL) {
        v = fr;
    } else {
        float mk = mask[(((size_t)b * (NSTEP - NIL) + (t - NIL)) * NHW + p) * NCH + c];
        float xg = xgen[(size_t)c * PIX + pix];
        v = mk * fr + (1.0f - mk) * xg;
    }
    actnet[(size_t)pix * 32 + c] = f2b(v);
}

// ---------------------------------------------------------------------------
// Write step output: d_out[b][t][h][w][c] = xgen_cc[c][pix].
// ---------------------------------------------------------------------------
__global__ void outwrite_kernel(const float* __restrict__ xgen, float* __restrict__ out, int t)
{
    int idx = blockIdx.x * 256 + threadIdx.x;
    if (idx >= PIX * NCH) return;
    int pix = idx >> 4;
    int c   = idx & 15;
    int b   = pix >> 10;
    int p   = pix & 1023;
    out[(((size_t)b * NSTEP + t) * NHW + p) * NCH + c] = xgen[(size_t)c * PIX + pix];
}

// ---------------------------------------------------------------------------
// Host orchestration.
// ---------------------------------------------------------------------------
extern "C" void kernel_launch(void* const* d_in, const int* in_sizes, int n_in,
                              void* d_out, int out_size, void* d_ws, size_t ws_size,
                              hipStream_t stream)
{
    const float* frames = (const float*)d_in[0];
    const float* mask   = (const float*)d_in[1];
    const float* Wx0    = (const float*)d_in[2];
    const float* Wx     = (const float*)d_in[3];
    const float* Wh     = (const float*)d_in[4];
    const float* Wm     = (const float*)d_in[5];
    const float* Wo     = (const float*)d_in[6];
    const float* Wc     = (const float*)d_in[7];
    const float* lnxg   = (const float*)d_in[8];
    const float* lnxb   = (const float*)d_in[9];
    const float* lnhg   = (const float*)d_in[10];
    const float* lnhb   = (const float*)d_in[11];
    const float* lnmg   = (const float*)d_in[12];
    const float* lnmb   = (const float*)d_in[13];
    const float* lnog   = (const float*)d_in[14];
    const float* lnob   = (const float*)d_in[15];
    const float* Wlast  = (const float*)d_in[16];
    float* out = (float*)d_out;

    char* ws = (char*)d_ws;
    size_t off = 0;
    auto alloc_f = [&](size_t n) { float*  p = (float*)(ws + off);  off += n * sizeof(float);  return p; };
    auto alloc_b = [&](size_t n) { bf16_t* p = (bf16_t*)(ws + off); off += n * sizeof(bf16_t); return p; };

    // f32 buffers (channel-major [C][PIX])
    float* xcc   = alloc_f((size_t)448 * PIX);
    float* hcc   = alloc_f((size_t)256 * PIX);
    float* mcc   = alloc_f((size_t)192 * PIX);
    float* occ   = alloc_f((size_t)64  * PIX);
    float* ccc   = alloc_f((size_t)64  * PIX);
    float* xgen  = alloc_f((size_t)64  * PIX);
    float* cst   = alloc_f((size_t)NL * 64 * PIX);   // c states, 4 layers
    float* memf  = alloc_f((size_t)64 * PIX);        // m state

    // bf16 weights: [tap][CoutP][CinP]
    bf16_t* wx0 = alloc_b((size_t)25 * 448 * 32);
    bf16_t* wxl[NL]; wxl[0] = nullptr;
    for (int l = 1; l < NL; ++l) wxl[l] = alloc_b((size_t)25 * 448 * 64);
    bf16_t* whl[NL]; for (int l = 0; l < NL; ++l) whl[l] = alloc_b((size_t)25 * 256 * 64);
    bf16_t* wml[NL]; for (int l = 0; l < NL; ++l) wml[l] = alloc_b((size_t)25 * 192 * 64);
    bf16_t* wol[NL]; for (int l = 0; l < NL; ++l) wol[l] = alloc_b((size_t)25 * 64 * 128);
    bf16_t* wcl[NL]; for (int l = 0; l < NL; ++l) wcl[l] = alloc_b((size_t)64 * 128);
    bf16_t* wlast = alloc_b((size_t)64 * 64);

    // bf16 activations, NHWC [PIX][CinP]
    bf16_t* actnet  = alloc_b((size_t)PIX * 32);
    bf16_t* acth[NL]; for (int l = 0; l < NL; ++l) acth[l] = alloc_b((size_t)PIX * 64);
    bf16_t* actm    = alloc_b((size_t)PIX * 64);
    bf16_t* actmem2 = alloc_b((size_t)PIX * 128);
    (void)ws_size; (void)in_sizes; (void)n_in; (void)out_size;

    // ---- zero the recurrent states (poison-proof, deterministic) ----
    hipMemsetAsync(cst,     0, (size_t)NL * 64 * PIX * sizeof(float),  stream);
    hipMemsetAsync(memf,    0, (size_t)64 * PIX * sizeof(float),       stream);
    hipMemsetAsync(acth[0], 0, (size_t)NL * PIX * 64 * sizeof(bf16_t), stream); // contiguous 4 layers
    hipMemsetAsync(actm,    0, (size_t)PIX * 64 * sizeof(bf16_t),      stream);

    // ---- weight prep ----
    auto prep = [&](const float* s, bf16_t* d, int Co, int Ci, int CoP, int CiP, int ks) {
        int total = ks * ks * CoP * CiP;
        wprep_kernel<<<(total + 255) / 256, 256, 0, stream>>>(s, d, Co, Ci, CoP, CiP, ks);
    };
    prep(Wx0, wx0, 448, 16, 448, 32, 5);
    for (int l = 1; l < NL; ++l) prep(Wx + (size_t)(l - 1) * 448 * 64 * 25, wxl[l], 448, 64, 448, 64, 5);
    for (int l = 0; l < NL; ++l) {
        prep(Wh + (size_t)l * 256 * 64 * 25, whl[l], 256, 64, 256, 64, 5);
        prep(Wm + (size_t)l * 192 * 64 * 25, wml[l], 192, 64, 192, 64, 5);
        prep(Wo + (size_t)l * 64 * 128 * 25, wol[l], 64, 128, 64, 128, 5);
        prep(Wc + (size_t)l * 64 * 128,      wcl[l], 64, 128, 64, 128, 1);
    }
    prep(Wlast, wlast, 16, 64, 64, 64, 1);

    auto gemm = [&](const bf16_t* a, const bf16_t* w, float* o, int CinP, int CoutP, int ks) {
        dim3 grid(PIX / 256, CoutP / 64);
        conv_gemm<<<grid, 256, 0, stream>>>(a, w, o, CinP, CoutP, ks);
    };
    auto ln = [&](float* x, const float* g, const float* b, int Cg) {
        ln_kernel<<<NB, 1024, 0, stream>>>(x, g, b, Cg);
    };
    const int EW = (NHID * PIX + 255) / 256;       // elementwise grid (64ch)
    const int EWC = (PIX * NCH + 255) / 256;       // elementwise grid (16ch)

    // ---- time loop ----
    for (int t = 0; t < NSTEP; ++t) {
        net_kernel<<<EWC, 256, 0, stream>>>(frames, mask, xgen, actnet, t);

        for (int i = 0; i < NL; ++i) {
            const bf16_t* axin = (i == 0) ? actnet : acth[i - 1];
            int cinX = (i == 0) ? 32 : 64;
            const bf16_t* wx = (i == 0) ? wx0 : wxl[i];

            gemm(axin, wx, xcc, cinX, 448, 5);
            ln(xcc, lnxg + (size_t)i * 448 * NHW, lnxb + (size_t)i * 448 * NHW, 448);

            gemm(acth[i], whl[i], hcc, 64, 256, 5);
            ln(hcc, lnhg + (size_t)i * 256 * NHW, lnhb + (size_t)i * 256 * NHW, 256);

            gemm(actm, wml[i], mcc, 64, 192, 5);
            ln(mcc, lnmg + (size_t)i * 192 * NHW, lnmb + (size_t)i * 192 * NHW, 192);

            gates_kernel<<<EW, 256, 0, stream>>>(xcc, hcc, mcc,
                cst + (size_t)i * 64 * PIX, memf, actm, actmem2);

            gemm(actmem2, wol[i], occ, 128, 64, 5);
            ln(occ, lnog + (size_t)i * 64 * NHW, lnob + (size_t)i * 64 * NHW, 64);

            gemm(actmem2, wcl[i], ccc, 128, 64, 1);

            outgate_kernel<<<EW, 256, 0, stream>>>(xcc, hcc, occ, ccc, acth[i]);
        }

        gemm(acth[NL - 1], wlast, xgen, 64, 64, 1);
        outwrite_kernel<<<EWC, 256, 0, stream>>>(xgen, out, t);
    }
}
